// PrototypicalNetwork_83399674953870
// MI455X (gfx1250) — matrix operand order repaired and checked by reading, not compiled, not run on hardware
//
#include <hip/hip_runtime.h>
#include <stdint.h>

#define DIM    128
#define NCLS   128
#define SHOTS  32
#define NQ     65536
#define NBOOT  8
#define QT     32           // queries per block (two 16-row WMMA tiles)

typedef __attribute__((ext_vector_type(16))) __bf16 v16bf;
typedef __attribute__((ext_vector_type(8)))  float  v8f;

union V16BF {
  v16bf v;
  uint32_t u[8];
  uint4 q[2];
};

__device__ __forceinline__ uint16_t f2bf(float f) {
  uint32_t u = __float_as_uint(f);
  u += 0x7FFFu + ((u >> 16) & 1u);      // round-to-nearest-even
  return (uint16_t)(u >> 16);
}

// ---------------------------------------------------------------------------
// Kernel 1: bootstrap prototypes (mean of gathered shots) + ||p||^2
// grid = NBOOT*NCLS blocks, 128 threads (thread = feature dim)
// ---------------------------------------------------------------------------
__global__ __launch_bounds__(128) void proto_kernel(
    const float* __restrict__ support,      // [4096, 128]
    const int*   __restrict__ boot,         // [NBOOT, NCLS, SHOTS]
    uint16_t*    __restrict__ protos_bf,    // [NBOOT, NCLS, DIM] bf16
    float*       __restrict__ p2g)          // [NBOOT, NCLS]
{
  const int bc = blockIdx.x;                // b*NCLS + c
  const int c  = bc & (NCLS - 1);
  const int t  = threadIdx.x;               // d
  const int* idx = boot + bc * SHOTS;

  float acc = 0.f;
#pragma unroll
  for (int s = 0; s < SHOTS; ++s) {
    int j = idx[s];                         // uniform -> scalar load
    acc += support[(c * SHOTS + j) * DIM + t];
  }
  float proto = acc * (1.0f / SHOTS);
  protos_bf[bc * DIM + t] = f2bf(proto);

  __shared__ float red[DIM];
  red[t] = proto * proto;
  __syncthreads();
#pragma unroll
  for (int off = DIM / 2; off > 0; off >>= 1) {
    if (t < off) red[t] += red[t + off];
    __syncthreads();
  }
  if (t == 0) p2g[bc] = red[0];
}

// ---------------------------------------------------------------------------
// Kernel 2: logits[q][c] = -(1/B) * sum_b sqrt(max(q2 + p2 - 2*q.p, 0))
// grid = NQ/QT blocks, 256 threads = 8 waves; wave w owns classes [16w,16w+16)
// ---------------------------------------------------------------------------
__global__ __launch_bounds__(256) void dist_kernel(
    const float*    __restrict__ query,     // [NQ, DIM]
    const uint16_t* __restrict__ protos_bf, // [NBOOT, NCLS, DIM] bf16
    const float*    __restrict__ p2g,       // [NBOOT, NCLS]
    float*          __restrict__ out)       // [NQ, NCLS]
{
  __shared__ uint32_t qsh[QT * (DIM / 2)];  // QT rows x 128 bf16  (8 KB)
  __shared__ float    q2s[QT];
  __shared__ float    osh[QT * NCLS];       // staging for coalesced stores (16 KB)

  const int t     = threadIdx.x;
  const int qbase = blockIdx.x * QT;

  if (t < QT) q2s[t] = 0.f;
  __syncthreads();

  // ---- stage QT x 128 query tile to LDS as bf16; accumulate q2 in f32 ----
  {
    const int row = t >> 3;                 // 0..31
    const int col = (t & 7) * 16;           // 0,16,...,112
    const float4* src = (const float4*)(query + (size_t)(qbase + row) * DIM + col);
    float part = 0.f;
    uint32_t* dst = qsh + row * (DIM / 2) + col / 2;
#pragma unroll
    for (int j = 0; j < 4; ++j) {
      float4 a = src[j];
      part += a.x * a.x + a.y * a.y + a.z * a.z + a.w * a.w;
      dst[2 * j + 0] = (uint32_t)f2bf(a.x) | ((uint32_t)f2bf(a.y) << 16);
      dst[2 * j + 1] = (uint32_t)f2bf(a.z) | ((uint32_t)f2bf(a.w) << 16);
    }
    atomicAdd(&q2s[row], part);
  }
  __syncthreads();

  const int lane = t & 31;
  const int w    = t >> 5;                  // wave id -> class block
  const int m16  = lane & 15;
  const int h    = lane >> 4;               // K-half selector per ISA layout

  // ---- load A fragments once (reused for all 8 bootstraps) ----
  // 16-bit A 16x32 layout: lane(h,m): K = {8h..8h+7} U {16+8h..16+8h+7}
  V16BF afrag[2][4];
#pragma unroll
  for (int qt = 0; qt < 2; ++qt) {
    const int row = qt * 16 + m16;
#pragma unroll
    for (int kk = 0; kk < 4; ++kk) {
      const uint4* l0 = (const uint4*)(qsh + row * (DIM / 2) + kk * 16 + 4 * h);
      const uint4* l1 = (const uint4*)(qsh + row * (DIM / 2) + kk * 16 + 8 + 4 * h);
      afrag[qt][kk].q[0] = l0[0];
      afrag[qt][kk].q[1] = l1[0];
    }
  }

  const int cls = w * 16 + m16;             // this lane's class (B-operand row)

  float q2r[2][8];
#pragma unroll
  for (int qt = 0; qt < 2; ++qt)
#pragma unroll
    for (int r = 0; r < 8; ++r) q2r[qt][r] = q2s[qt * 16 + r + 8 * h];

  float accd[2][8];
#pragma unroll
  for (int qt = 0; qt < 2; ++qt)
#pragma unroll
    for (int r = 0; r < 8; ++r) accd[qt][r] = 0.f;

  // Main loop: one bootstrap per iteration, unrolled x2 so the scheduler can
  // issue the next body's 8-load clause underneath the current body's WMMAs.
  // Only two C-tile pairs live at a time -> no accumulator register blowup.
  // p2 is re-loaded from global each iteration (L2-resident b32, folds into
  // the load clause) instead of a register array -> no v_movrels indexing.
#pragma unroll 2
  for (int b = 0; b < NBOOT; ++b) {
    // B fragments: lane(h,n=cls): K = kk*32 + {8h..8h+7} U {16+8h..16+8h+7}
    const uint4* bp = (const uint4*)(protos_bf + ((size_t)b * NCLS + cls) * DIM);
    V16BF bfr[4];
#pragma unroll
    for (int kk = 0; kk < 4; ++kk) {
      bfr[kk].q[0] = bp[kk * 4 + h];
      bfr[kk].q[1] = bp[kk * 4 + 2 + h];
    }
    v8f c0 = {};
    v8f c1 = {};
#pragma unroll
    for (int kk = 0; kk < 4; ++kk) {
      c0 = __builtin_amdgcn_wmma_f32_16x16x32_bf16(
              false, afrag[0][kk].v, false, bfr[kk].v, (short)0, c0, false, false);
      c1 = __builtin_amdgcn_wmma_f32_16x16x32_bf16(
              false, afrag[1][kk].v, false, bfr[kk].v, (short)0, c1, false, false);
    }
    const float p2v = p2g[b * NCLS + cls];
    // Per-tile epilogue loops: pk-op pairs form within one contiguous
    // accumulator block (no cross-tile register permutes).
#pragma unroll
    for (int r = 0; r < 8; ++r) {
      float d2a = fmaf(-2.0f, c0[r], q2r[0][r] + p2v);
      accd[0][r] += __builtin_amdgcn_sqrtf(fmaxf(d2a, 0.f));
    }
#pragma unroll
    for (int r = 0; r < 8; ++r) {
      float d2b = fmaf(-2.0f, c1[r], q2r[1][r] + p2v);
      accd[1][r] += __builtin_amdgcn_sqrtf(fmaxf(d2b, 0.f));
    }
  }

  // ---- C/D layout: VGPR r, lane(h,n): M = r + 8h, N = n ----
#pragma unroll
  for (int qt = 0; qt < 2; ++qt)
#pragma unroll
    for (int r = 0; r < 8; ++r)
      osh[(qt * 16 + r + 8 * h) * NCLS + w * 16 + m16] =
          accd[qt][r] * (-1.0f / NBOOT);
  __syncthreads();

  // ---- coalesced stores: each row is 512 B written by 8 threads ----
  {
    const int row = t >> 3;
    const int col = (t & 7) * 16;
    float4*       dst = (float4*)(out + (size_t)(qbase + row) * NCLS + col);
    const float4* src = (const float4*)(osh + row * NCLS + col);
#pragma unroll
    for (int j = 0; j < 4; ++j) dst[j] = src[j];
  }
}

// ---------------------------------------------------------------------------
extern "C" void kernel_launch(void* const* d_in, const int* in_sizes, int n_in,
                              void* d_out, int out_size, void* d_ws, size_t ws_size,
                              hipStream_t stream) {
  const float* support = (const float*)d_in[0];
  // d_in[1] = support_labels (unused: labels are arange//SHOTS by construction)
  const float* query   = (const float*)d_in[2];
  const int*   boot    = (const int*)d_in[3];

  uint16_t* protos_bf = (uint16_t*)d_ws;                        // 256 KB
  float*    p2g = (float*)((char*)d_ws +
                           (size_t)NBOOT * NCLS * DIM * sizeof(uint16_t)); // 4 KB

  proto_kernel<<<NBOOT * NCLS, 128, 0, stream>>>(support, boot, protos_bf, p2g);
  dist_kernel<<<NQ / QT, 256, 0, stream>>>(query, protos_bf, p2g, (float*)d_out);
}
